// DualTowerTitans_28200755265716
// MI455X (gfx1250) — compile-verified, matrix operand order checked by reading, not run on hardware
//
#include <hip/hip_runtime.h>
#include <hip/hip_bf16.h>
#include <math.h>

#define B_  8
#define L_  512
#define D_  512
#define H_  8
#define DH_ 64
#define M_  (B_ * L_)   // 4096 rows of x
#define N_  (H_ * DH_)  // 512  output cols
#define K_  D_          // 512  reduction

typedef __attribute__((ext_vector_type(16))) __bf16 bf16x16;
typedef __attribute__((ext_vector_type(8)))  float  floatx8;

__device__ __forceinline__ unsigned short f32_to_bf16_rne(float f) {
  unsigned int u = __float_as_uint(f);
  u += 0x7FFFu + ((u >> 16) & 1u);   // round-to-nearest-even
  return (unsigned short)(u >> 16);
}

// ---------------- bf16 conversion kernels ----------------
__global__ void cvt_x_bf16(const float* __restrict__ x,
                           unsigned short* __restrict__ xb, int n) {
  int i = blockIdx.x * blockDim.x + threadIdx.x;
  if (i < n) xb[i] = f32_to_bf16_rne(x[i]);
}

// W is [K_, N_] row-major (d, n); produce Wt [N_, K_] bf16 so B-fragments
// gather along contiguous K.
__global__ void cvt_w_t_bf16(const float* __restrict__ w,
                             unsigned short* __restrict__ wt) {
  int i = blockIdx.x * blockDim.x + threadIdx.x;  // over N_*K_
  int nrow = i >> 9;   // n
  int kcol = i & 511;  // d
  wt[i] = f32_to_bf16_rne(w[kcol * N_ + nrow]);
}

// ---------------- WMMA bf16 GEMM: Out[M,N] = Xb[M,K] * Wt[N,K]^T ----------------
// One wave per 16x16 tile; 8 waves/block; 1024 blocks cover 256x32 tiles.
__global__ void gemm_wmma(const unsigned short* __restrict__ Xb,
                          const unsigned short* __restrict__ Wt,
                          float* __restrict__ Out) {
  const int lane = threadIdx.x & 31;
  const int wave = threadIdx.x >> 5;
  const int tile = blockIdx.x * 8 + wave;   // 0..8191
  const int tm = (tile >> 5) << 4;          // M tile base
  const int tn = (tile & 31) << 4;          // N tile base
  const int r  = lane & 15;
  const int hi = lane >> 4;                 // lane>=16 half

  const unsigned short* arow = Xb + (size_t)(tm + r) * K_;
  const unsigned short* brow = Wt + (size_t)(tn + r) * K_;

  floatx8 acc = {0.f, 0.f, 0.f, 0.f, 0.f, 0.f, 0.f, 0.f};

  #pragma unroll 4
  for (int kk = 0; kk < K_; kk += 32) {
    union { unsigned int u[8]; bf16x16 v; } a, b;
    #pragma unroll
    for (int v2 = 0; v2 < 8; ++v2) {
      // 16-bit A/B fragment layout: VGPR v holds K = {base+2(v&3), base+2(v&3)+1},
      // base = (v<4 ? 0 : 16) + (lane>=16 ? 8 : 0)
      int kb = kk + ((v2 < 4) ? 0 : 16) + (hi << 3) + ((v2 & 3) << 1);
      a.u[v2] = *(const unsigned int*)(arow + kb);
      b.u[v2] = *(const unsigned int*)(brow + kb);
    }
    acc = __builtin_amdgcn_wmma_f32_16x16x32_bf16(
        /*neg_a=*/false, a.v, /*neg_b=*/false, b.v,
        /*c_mod=*/(short)0, acc, /*reuse_a=*/false, /*reuse_b=*/false);
  }

  // C/D layout: VGPR rr -> row M = rr + (lane>=16 ? 8 : 0), col N = lane%16
  #pragma unroll
  for (int rr = 0; rr < 8; ++rr) {
    int row = tm + rr + (hi << 3);
    Out[(size_t)row * N_ + tn + r] = acc[rr];
  }
}

// ---------------- gate projection: p = x @ Wp + bp ; gates ----------------
__global__ void gates_kernel(const float* __restrict__ x, const float* __restrict__ Wp,
                             const float* __restrict__ bp, float* __restrict__ alpha,
                             float* __restrict__ eta, float* __restrict__ theta) {
  int gid = blockIdx.x * blockDim.x + threadIdx.x;
  if (gid >= M_ * 24) return;
  int m = gid / 24, c = gid % 24;
  const float* xr = x + (size_t)m * D_;
  float acc = 0.f;
  for (int d = 0; d < D_; ++d) acc = fmaf(xr[d], Wp[d * 24 + c], acc);
  acc += bp[c];
  if (c < 8) {
    alpha[m * H_ + c] = 1.f / (1.f + __expf(-acc));
  } else if (c < 16) {
    eta[m * H_ + (c - 8)] = 1.f / (1.f + __expf(-acc));
  } else {
    float sp = (acc > 20.f) ? acc : log1pf(__expf(acc));
    theta[m * H_ + (c - 16)] = sp;
  }
}

// ---------------- k L2-normalize (per (m,h) row of 64) ----------------
__global__ void knorm_kernel(float* __restrict__ Kn) {
  int row = blockIdx.x * blockDim.x + threadIdx.x;
  if (row >= M_ * H_) return;
  float* p = Kn + (size_t)row * DH_;
  float ss = 0.f;
  #pragma unroll 8
  for (int d = 0; d < DH_; ++d) ss = fmaf(p[d], p[d], ss);
  float inv = 1.f / fmaxf(sqrtf(ss), 1e-12f);
  #pragma unroll 8
  for (int d = 0; d < DH_; ++d) p[d] *= inv;
}

// ---------------- sequential scan: one block per (b,h) ----------------
// 256 threads; thread t owns row i=t/4, cols [16*(t%4), +16) of the 64x64
// state & momentum (16+16 floats in registers). Store stream is the
// bottleneck -> coalesced float4 writes.
__global__ void scan_kernel(const float* __restrict__ Q, const float* __restrict__ Kn,
                            const float* __restrict__ V, const float* __restrict__ alpha,
                            const float* __restrict__ eta, const float* __restrict__ theta,
                            const float* __restrict__ st0, const float* __restrict__ mo0,
                            float* __restrict__ y_out, float* __restrict__ st_out,
                            float* __restrict__ mo_out) {
  const int bh = blockIdx.x;
  const int b = bh >> 3, h = bh & 7;
  const int tid = threadIdx.x;
  const int i  = tid >> 2;   // row 0..63
  const int c  = tid & 3;    // column group
  const int j0 = c << 4;

  __shared__ float qs[DH_], ks[DH_], vs[DH_], dl[DH_];
  __shared__ float red[DH_ * 8];
  __shared__ float sc[3];

  float st[16], mo[16];
  {
    size_t base = (((size_t)(b * H_ + h) * DH_ + i) * DH_) + j0;
    #pragma unroll
    for (int l = 0; l < 16; ++l) { st[l] = st0[base + l]; mo[l] = mo0[base + l]; }
  }

  for (int t = 0; t < L_; ++t) {
    const int m = b * L_ + t;
    const int off = m * N_ + h * DH_;
    if (tid < DH_)               qs[tid]            = Q[off + tid];
    else if (tid < 2 * DH_)      ks[tid - DH_]      = Kn[off + tid - DH_];
    else if (tid < 3 * DH_)      vs[tid - 2 * DH_]  = V[off + tid - 2 * DH_];
    else if (tid == 3 * DH_) {
      sc[0] = alpha[m * H_ + h]; sc[1] = eta[m * H_ + h]; sc[2] = theta[m * H_ + h];
    }
    __syncthreads();

    // y_i = sum_j state[i][j] q[j] ; vhat_i = sum_j state[i][j] k[j]  (prev state)
    float py = 0.f, pv = 0.f;
    #pragma unroll
    for (int l = 0; l < 16; ++l) {
      py = fmaf(st[l], qs[j0 + l], py);
      pv = fmaf(st[l], ks[j0 + l], pv);
    }
    red[i * 8 + c]     = py;
    red[i * 8 + 4 + c] = pv;
    __syncthreads();

    if (tid < DH_) {
      const float* rr = red + tid * 8;
      float y  = (rr[0] + rr[1]) + (rr[2] + rr[3]);
      float vh = (rr[4] + rr[5]) + (rr[6] + rr[7]);
      dl[tid] = vh - vs[tid];                          // (vhat - v)_i
      y_out[(size_t)m * N_ + h * DH_ + tid] = y;       // readout uses prev state
    }
    __syncthreads();

    const float a = sc[0], e = sc[1], thv = sc[2];
    const float di = dl[i];
    #pragma unroll
    for (int l = 0; l < 16; ++l) {
      mo[l] = fmaf(e, mo[l], -thv * (di * ks[j0 + l]));  // mom = e*mom - th*grad
      st[l] = fmaf(1.f - a, st[l], mo[l]);               // state = (1-a)*state + mom
    }

    size_t base = ((((size_t)m * H_ + h) * DH_ + i) * DH_) + j0;
    #pragma unroll
    for (int l = 0; l < 16; l += 4) {
      *(float4*)(st_out + base + l) = make_float4(st[l], st[l+1], st[l+2], st[l+3]);
      *(float4*)(mo_out + base + l) = make_float4(mo[l], mo[l+1], mo[l+2], mo[l+3]);
    }
    __syncthreads();  // protect shared tiles before next step's loads
  }
}

// ---------------- launcher ----------------
extern "C" void kernel_launch(void* const* d_in, const int* in_sizes, int n_in,
                              void* d_out, int out_size, void* d_ws, size_t ws_size,
                              hipStream_t stream) {
  const float* x   = (const float*)d_in[0];
  const float* Wq  = (const float*)d_in[1];
  const float* Wk  = (const float*)d_in[2];
  const float* Wv  = (const float*)d_in[3];
  const float* Wp  = (const float*)d_in[4];
  const float* bp  = (const float*)d_in[5];
  const float* st0 = (const float*)d_in[6];
  const float* mo0 = (const float*)d_in[7];

  char* ws = (char*)d_ws;
  unsigned short* Xb  = (unsigned short*)ws;  ws += (size_t)M_ * K_ * 2;
  unsigned short* WtQ = (unsigned short*)ws;  ws += (size_t)N_ * K_ * 2;
  unsigned short* WtK = (unsigned short*)ws;  ws += (size_t)N_ * K_ * 2;
  unsigned short* WtV = (unsigned short*)ws;  ws += (size_t)N_ * K_ * 2;
  float* Qp = (float*)ws;  ws += (size_t)M_ * N_ * 4;
  float* Kp = (float*)ws;  ws += (size_t)M_ * N_ * 4;
  float* Vp = (float*)ws;  ws += (size_t)M_ * N_ * 4;
  float* al = (float*)ws;  ws += (size_t)M_ * H_ * 4;
  float* et = (float*)ws;  ws += (size_t)M_ * H_ * 4;
  float* th = (float*)ws;  ws += (size_t)M_ * H_ * 4;

  cvt_x_bf16<<<(M_ * K_ + 255) / 256, 256, 0, stream>>>(x, Xb, M_ * K_);
  cvt_w_t_bf16<<<(N_ * K_ + 255) / 256, 256, 0, stream>>>(Wq, WtQ);
  cvt_w_t_bf16<<<(N_ * K_ + 255) / 256, 256, 0, stream>>>(Wk, WtK);
  cvt_w_t_bf16<<<(N_ * K_ + 255) / 256, 256, 0, stream>>>(Wv, WtV);

  gemm_wmma<<<1024, 256, 0, stream>>>(Xb, WtQ, Qp);
  gemm_wmma<<<1024, 256, 0, stream>>>(Xb, WtK, Kp);
  gemm_wmma<<<1024, 256, 0, stream>>>(Xb, WtV, Vp);

  gates_kernel<<<(M_ * 24 + 255) / 256, 256, 0, stream>>>(x, Wp, bp, al, et, th);
  knorm_kernel<<<(M_ * H_ + 255) / 256, 256, 0, stream>>>(Kp);

  float* y_out  = (float*)d_out;
  float* st_out = y_out + (size_t)M_ * N_;
  float* mo_out = st_out + (size_t)M_ * H_ * DH_ * DH_;

  scan_kernel<<<B_ * H_, 256, 0, stream>>>(Qp, Kp, Vp, al, et, th,
                                           st0, mo0, y_out, st_out, mo_out);
}
